// SelfAttention_6768868458538
// MI455X (gfx1250) — matrix operand order repaired and checked by reading, not compiled, not run on hardware
//
#include <hip/hip_runtime.h>
#include <hip/hip_bf16.h>

// ---------------------------------------------------------------------------
// Self-attention for MI455X (gfx1250): bf16 WMMA GEMMs, double-buffered LDS
// stages with async global->LDS copies (ASYNCcnt) where operands are bf16.
//   Q = x@Wq + bq ; K = x@Wk + bk ; V = x@Wv + bv        (bf16 in ws)
//   S = Q@K^T / sqrt(32)  -> f32 directly into d_out attn_weights region
//   P = softmax_rows(S)   -> f32 in place + bf16 copy in ws
//   O = P@V               -> f32 into d_out attn_output region
// ---------------------------------------------------------------------------

#if defined(__has_builtin)
#if __has_builtin(__builtin_amdgcn_global_load_async_to_lds_b128) && \
    __has_builtin(__builtin_amdgcn_s_wait_asynccnt)
#define HAVE_ASYNC_LDS 1
#endif
#endif
#ifndef HAVE_ASYNC_LDS
#define HAVE_ASYNC_LDS 0
#endif

typedef __attribute__((ext_vector_type(16))) __bf16 v16bf;
typedef __attribute__((ext_vector_type(8)))  float  v8f;

#if HAVE_ASYNC_LDS
typedef int v4i __attribute__((__vector_size__(4 * sizeof(int))));
typedef __attribute__((address_space(1))) v4i* gas1_p;   // non-const: matches builtin proto
typedef __attribute__((address_space(3))) v4i* las3_p;
#endif

#define TILE_M 128
#define TILE_N 128
#define TILE_K 32
#define LDS_K  40   // padded LDS row stride (bf16 elems): 80B rows, conflict-free b128

__device__ __forceinline__ unsigned short f2bf(float f) {
    union { float f; unsigned u; } x; x.f = f;
    unsigned r = x.u + 0x7FFFu + ((x.u >> 16) & 1u);   // round-to-nearest-even
    return (unsigned short)(r >> 16);
}

// 16B global -> LDS copy: async-to-LDS on gfx1250 toolchains that expose it,
// else a plain VGPR-mediated copy (compiler handles dscnt waits at barriers).
__device__ __forceinline__ void copy16_g2l(const void* g, void* l) {
#if HAVE_ASYNC_LDS
    // Generic LDS pointer = aperture | lds_offset: low 32 bits are the AS3 address.
    __builtin_amdgcn_global_load_async_to_lds_b128(
        (gas1_p)(unsigned long long)g,
        (las3_p)(unsigned)(unsigned long long)l, 0, 0);
#else
    *(uint4*)l = *(const uint4*)g;
#endif
}

template<int N>
__device__ __forceinline__ void wait_async() {
#if HAVE_ASYNC_LDS
    __builtin_amdgcn_s_wait_asynccnt(N);
#endif
}

// A: [M,K] row-major (f32 or bf16).  B: if B_NT, [N,K] bf16 (already B^T form);
// else [K,N] (f32 or bf16), transposed into LDS.  Output: f32 (scaled) or bf16 (+bias).
template<bool A_F32, bool B_F32, bool B_NT, bool BIAS, bool OUT_BF16>
__global__ __launch_bounds__(256)
void gemm_wmma(const void* __restrict__ Ag, const void* __restrict__ Bg,
               const float* __restrict__ bias,
               float* __restrict__ Co, unsigned short* __restrict__ Cb,
               int K, int lda, int ldb, int ldc, float scale)
{
    __shared__ __align__(16) unsigned short As[2][TILE_M * LDS_K];
    __shared__ __align__(16) unsigned short Bs[2][TILE_N * LDS_K];

    const int tileM = blockIdx.y * TILE_M;
    const int tileN = blockIdx.x * TILE_N;
    const int tid   = threadIdx.x;
    const int lane  = tid & 31;
    const int wave  = tid >> 5;           // 8 waves
    const int waveM = (wave >> 2) * 64;   // 2 wave-rows
    const int waveN = (wave & 3) * 32;    // 4 wave-cols
    const int h     = lane >> 4;          // lane half (0/1)
    const int l16   = lane & 15;

    // async issues per thread per stage (bf16 direct-copy operands only)
    constexpr int ASY = (A_F32 ? 0 : 2) + (B_NT ? 2 : 0);

    auto load_stage = [&](int k0, int buf) {
        // ---------------- A tile (128 x 32) -> LDS bf16 ----------------
        if constexpr (A_F32) {
            const float* A = (const float*)Ag;
            const int r0 = tid >> 3, c = (tid & 7) * 4;
            #pragma unroll
            for (int p = 0; p < 4; ++p) {
                const int row = r0 + 32 * p;
                const float4 v = *(const float4*)&A[(size_t)(tileM + row) * lda + k0 + c];
                unsigned long long pk = (unsigned long long)f2bf(v.x)
                                      | ((unsigned long long)f2bf(v.y) << 16)
                                      | ((unsigned long long)f2bf(v.z) << 32)
                                      | ((unsigned long long)f2bf(v.w) << 48);
                *(unsigned long long*)&As[buf][row * LDS_K + c] = pk;
            }
        } else {
            const unsigned short* A = (const unsigned short*)Ag;
            const int r0 = tid >> 2, c = (tid & 3) * 8;
            #pragma unroll
            for (int p = 0; p < 2; ++p) {
                const int row = r0 + 64 * p;
                copy16_g2l(&A[(size_t)(tileM + row) * lda + k0 + c],
                           &As[buf][row * LDS_K + c]);
            }
        }
        // ---------------- B tile -> LDS as B^T [n][k] bf16 ----------------
        if constexpr (B_NT) {             // B already [N,K] bf16: direct rows
            const unsigned short* B = (const unsigned short*)Bg;
            const int r0 = tid >> 2, c = (tid & 3) * 8;
            #pragma unroll
            for (int p = 0; p < 2; ++p) {
                const int row = r0 + 64 * p;
                copy16_g2l(&B[(size_t)(tileN + row) * ldb + k0 + c],
                           &Bs[buf][row * LDS_K + c]);
            }
        } else if constexpr (B_F32) {     // B [K,N] f32: transpose while converting
            const float* B = (const float*)Bg;
            const int n4 = (tid & 31) * 4, kr = tid >> 5;
            #pragma unroll
            for (int p = 0; p < 4; ++p) {
                const int kk = kr + 8 * p;
                const float4 v = *(const float4*)&B[(size_t)(k0 + kk) * ldb + tileN + n4];
                Bs[buf][(n4 + 0) * LDS_K + kk] = f2bf(v.x);
                Bs[buf][(n4 + 1) * LDS_K + kk] = f2bf(v.y);
                Bs[buf][(n4 + 2) * LDS_K + kk] = f2bf(v.z);
                Bs[buf][(n4 + 3) * LDS_K + kk] = f2bf(v.w);
            }
        } else {                          // B [K,N] bf16: transpose
            const unsigned short* B = (const unsigned short*)Bg;
            const int n8 = (tid & 15) * 8, kr = tid >> 4;
            #pragma unroll
            for (int p = 0; p < 2; ++p) {
                const int kk = kr + 16 * p;
                const uint4 v = *(const uint4*)&B[(size_t)(k0 + kk) * ldb + tileN + n8];
                const unsigned short* e = (const unsigned short*)&v;
                #pragma unroll
                for (int i = 0; i < 8; ++i)
                    Bs[buf][(n8 + i) * LDS_K + kk] = e[i];
            }
        }
    };

    v8f acc[4][2] = {};
    const int nst = K / TILE_K;

    load_stage(0, 0);                     // prologue: stage 0 in flight

    for (int s = 0; s < nst; ++s) {
        const int cur = s & 1;
        if (s + 1 < nst) {
            load_stage((s + 1) * TILE_K, cur ^ 1);    // next stage behind compute
            if constexpr (ASY > 0) wait_async<ASY>(); // stage s landed in LDS
        } else {
            if constexpr (ASY > 0) wait_async<0>();
        }
        __syncthreads();

        // ---------------- fragments per documented WMMA VGPR layout ----------------
        v16bf afr[4];
        #pragma unroll
        for (int mt = 0; mt < 4; ++mt) {
            const unsigned short* sp = &As[cur][(waveM + mt * 16 + l16) * LDS_K + h * 8];
            uint4* ap = (uint4*)&afr[mt];
            ap[0] = *(const uint4*)sp;          // K = h*8 .. h*8+7
            ap[1] = *(const uint4*)(sp + 16);   // K = 16+h*8 .. +7
        }
        v16bf bfr[2];
        #pragma unroll
        for (int nt = 0; nt < 2; ++nt) {
            const unsigned short* sp = &Bs[cur][(waveN + nt * 16 + l16) * LDS_K + h * 8];
            uint4* bp = (uint4*)&bfr[nt];
            bp[0] = *(const uint4*)sp;          // K = h*8 .. h*8+7
            bp[1] = *(const uint4*)(sp + 16);   // K = 16+h*8 .. +7
        }
        #pragma unroll
        for (int mt = 0; mt < 4; ++mt)
            #pragma unroll
            for (int nt = 0; nt < 2; ++nt)
                acc[mt][nt] = __builtin_amdgcn_wmma_f32_16x16x32_bf16(
                    false, afr[mt], false, bfr[nt], (short)0, acc[mt][nt], false, false);
        __syncthreads();
    }

    // ------------------ epilogue: C/D layout -> global ------------------
    #pragma unroll
    for (int mt = 0; mt < 4; ++mt) {
        #pragma unroll
        for (int nt = 0; nt < 2; ++nt) {
            const int col = tileN + waveN + nt * 16 + l16;
            #pragma unroll
            for (int j = 0; j < 8; ++j) {
                const int row = tileM + waveM + mt * 16 + h * 8 + j;
                float v = acc[mt][nt][j] * scale;
                if (BIAS) v += bias[col];
                if (OUT_BF16) Cb[(size_t)row * ldc + col] = f2bf(v);
                else          Co[(size_t)row * ldc + col] = v;
            }
        }
    }
}

// One block per row: softmax over 4096 f32 in place, bf16 copy to P.
__global__ __launch_bounds__(256)
void softmax_rows(float* __restrict__ S, unsigned short* __restrict__ P, int n)
{
    __shared__ float red[256];
    float* srow = S + (size_t)blockIdx.x * n;
    const int tid = threadIdx.x;

    float v[16];
    float m = -3.402823e38f;
    #pragma unroll
    for (int i = 0; i < 16; ++i) { v[i] = srow[tid + i * 256]; m = fmaxf(m, v[i]); }
    red[tid] = m; __syncthreads();
    for (int s = 128; s > 0; s >>= 1) {
        if (tid < s) red[tid] = fmaxf(red[tid], red[tid + s]);
        __syncthreads();
    }
    m = red[0]; __syncthreads();

    float sum = 0.f;
    #pragma unroll
    for (int i = 0; i < 16; ++i) { v[i] = __expf(v[i] - m); sum += v[i]; }
    red[tid] = sum; __syncthreads();
    for (int s = 128; s > 0; s >>= 1) {
        if (tid < s) red[tid] += red[tid + s];
        __syncthreads();
    }
    const float inv = 1.0f / red[0];

    unsigned short* prow = P + (size_t)blockIdx.x * n;
    #pragma unroll
    for (int i = 0; i < 16; ++i) {
        const float w = v[i] * inv;
        srow[tid + i * 256] = w;
        prow[tid + i * 256] = f2bf(w);
    }
}

extern "C" void kernel_launch(void* const* d_in, const int* in_sizes, int n_in,
                              void* d_out, int out_size, void* d_ws, size_t ws_size,
                              hipStream_t stream) {
    const float* x  = (const float*)d_in[0];
    const float* Wq = (const float*)d_in[1];
    const float* bq = (const float*)d_in[2];
    const float* Wk = (const float*)d_in[3];
    const float* bk = (const float*)d_in[4];
    const float* Wv = (const float*)d_in[5];
    const float* bv = (const float*)d_in[6];

    const int N = 4096, Din = 1024, Dp = 1024;
    const float inv_dk = 0.17677669529663687f;   // 1/sqrt(32)

    float* out_attn = (float*)d_out;                       // [N, Dp]
    float* out_w    = out_attn + (size_t)N * Dp;           // [N, N]

    unsigned short* Qb = (unsigned short*)d_ws;            // bf16 [N, Dp]
    unsigned short* Kb = Qb + (size_t)N * Dp;
    unsigned short* Vb = Kb + (size_t)N * Dp;
    unsigned short* Pb = Vb + (size_t)N * Dp;              // bf16 [N, N]

    dim3 blk(256);
    dim3 gProj(Dp / TILE_N, N / TILE_M);                   // (8, 32)
    dim3 gScores(N / TILE_N, N / TILE_M);                  // (32, 32)

    // QKV projections: f32 A, f32 B (transpose), +bias, bf16 out
    gemm_wmma<true, true, false, true, true><<<gProj, blk, 0, stream>>>(
        x, Wq, bq, nullptr, Qb, Din, Din, Dp, Dp, 1.0f);
    gemm_wmma<true, true, false, true, true><<<gProj, blk, 0, stream>>>(
        x, Wk, bk, nullptr, Kb, Din, Din, Dp, Dp, 1.0f);
    gemm_wmma<true, true, false, true, true><<<gProj, blk, 0, stream>>>(
        x, Wv, bv, nullptr, Vb, Din, Din, Dp, Dp, 1.0f);

    // Scores: bf16 A, bf16 B already [N,K] (K-matrix), scaled f32 -> d_out weights region
    gemm_wmma<false, false, true, false, false><<<gScores, blk, 0, stream>>>(
        Qb, Kb, nullptr, out_w, nullptr, Dp, Dp, Dp, N, inv_dk);

    // Row softmax in place + bf16 copy
    softmax_rows<<<N, blk, 0, stream>>>(out_w, Pb, N);

    // Output: bf16 A (P), bf16 B [K,N] (V, transpose), f32 -> d_out attn region
    gemm_wmma<false, false, false, false, false><<<gProj, blk, 0, stream>>>(
        Pb, Vb, nullptr, out_attn, nullptr, N, N, Dp, Dp, 1.0f);
}
